// MaskedMultiHeadedSelfAttention_72129680769303
// MI455X (gfx1250) — matrix operand order repaired and checked by reading, not compiled, not run on hardware
//
#include <hip/hip_runtime.h>
#include <cstdint>
#include <cstddef>

#define DMODEL 1024
#define NHEADS 16
#define HDIM   64
#define BATCH  4
#define SEQ    2048
#define BT     (BATCH * SEQ)   // 8192

typedef __attribute__((ext_vector_type(16))) __bf16        v16bf;
typedef __attribute__((ext_vector_type(8)))  float         v8f;
typedef __attribute__((ext_vector_type(4)))  unsigned int  u32x4;

union Frag16 {
  v16bf v;
  u32x4 q[2];
  unsigned short s[16];
};

__device__ __forceinline__ unsigned short f2bf(float f) {
  unsigned int u = __float_as_uint(f);
  u += 0x7fffu + ((u >> 16) & 1u);   // round-to-nearest-even
  return (unsigned short)(u >> 16);
}

__device__ __forceinline__ v8f wmma_bf16(const Frag16& a, const Frag16& b, v8f c) {
  // D = A(16x32 bf16) * B(32x16 bf16) + C(16x16 f32)
  return __builtin_amdgcn_wmma_f32_16x16x32_bf16(false, a.v, false, b.v,
                                                 (short)0, c, false, false);
}

// A fragment: lane holds row M = lane%16; elements e=0..7 -> K=(lane/16)*8+e,
// e=8..15 -> K=16+(lane/16)*8+(e-8). Two contiguous 8xbf16 (16B) loads.
__device__ __forceinline__ void loadA(Frag16& f, const unsigned short* rowPtr, int hlf) {
  f.q[0] = *(const u32x4*)(rowPtr + hlf * 8);
  f.q[1] = *(const u32x4*)(rowPtr + 16 + hlf * 8);
}
// B fragment: lane holds column N = lane%16; e=0..7 -> K=(lane/16)*8+e,
// e=8..15 -> K=16+(lane/16)*8+(e-8) (same as A).
// colPtr points at row N of B^T (i.e. column N of B), K-major.
__device__ __forceinline__ void loadB(Frag16& f, const unsigned short* colPtr, int hlf) {
  const unsigned short* p = colPtr + hlf * 8;
  f.q[0] = *(const u32x4*)(p);
  f.q[1] = *(const u32x4*)(p + 16);
}

// ---------------- conversion kernels ----------------

__global__ void cvt_bf16_kernel(const float* __restrict__ in,
                                unsigned short* __restrict__ out, int n) {
  int i = blockIdx.x * blockDim.x + threadIdx.x;
  if (i < n) out[i] = f2bf(in[i]);
}

// in: [K rows, N cols] row-major fp32 -> out: [N, K] row-major bf16 (transpose)
__global__ void cvt_t_bf16_kernel(const float* __restrict__ in,
                                  unsigned short* __restrict__ out, int K, int N) {
  int i = blockIdx.x * blockDim.x + threadIdx.x;
  if (i < K * N) {
    int k = i / N, n = i % N;
    out[(size_t)n * K + k] = f2bf(in[i]);
  }
}

// ---------------- QKV projection GEMM ----------------
// C[8192,3072] = Xb[8192,1024] @ Wqkv + bias ; scatter into Q,K (B,H,T,Dh) and V^T (B,H,Dh,T)
// Each wave: 16 rows x 64 cols (4 N-tiles) -> A fragment reused 4x per k-step.

__global__ __launch_bounds__(128) void gemm_qkv_kernel(
    const unsigned short* __restrict__ Xb,
    const unsigned short* __restrict__ WT,      // [3072,1024] bf16 (Wqkv^T)
    const float* __restrict__ bias,             // [3072]
    unsigned short* __restrict__ Qb,
    unsigned short* __restrict__ Kb,
    unsigned short* __restrict__ Vb) {
  const int wave  = threadIdx.x >> 5;
  const int lane  = threadIdx.x & 31;
  const int col16 = lane & 15;
  const int hlf   = lane >> 4;
  const int row0  = blockIdx.y * 16;                   // token tile
  const int col0  = blockIdx.x * 256 + wave * 64;      // 64-wide output slab

  const unsigned short* ap0 = Xb + (size_t)(row0 + col16) * DMODEL;
  const unsigned short* bp0 = WT + (size_t)(col0 + col16) * DMODEL;   // N-tile 0
  const unsigned short* bp1 = bp0 + (size_t)16 * DMODEL;
  const unsigned short* bp2 = bp0 + (size_t)32 * DMODEL;
  const unsigned short* bp3 = bp0 + (size_t)48 * DMODEL;

  v8f acc0 = {}, acc1 = {}, acc2 = {}, acc3 = {};
  for (int k = 0; k < DMODEL; k += 32) {
    __builtin_prefetch(ap0 + k + 512, 0, 1);   // global_prefetch_b8, 1 KiB ahead
    __builtin_prefetch(bp0 + k + 512, 0, 1);
    Frag16 a, b;
    loadA(a, ap0 + k, hlf);
    loadB(b, bp0 + k, hlf); acc0 = wmma_bf16(a, b, acc0);
    loadB(b, bp1 + k, hlf); acc1 = wmma_bf16(a, b, acc1);
    loadB(b, bp2 + k, hlf); acc2 = wmma_bf16(a, b, acc2);
    loadB(b, bp3 + k, hlf); acc3 = wmma_bf16(a, b, acc3);
  }

  v8f accs[4] = {acc0, acc1, acc2, acc3};
#pragma unroll
  for (int nt = 0; nt < 4; ++nt) {
    const int   c     = col0 + nt * 16 + col16;
    const float bv    = bias[c];
    const int   which = c >> 10;       // 0=Q 1=K 2=V
    const int   dm    = c & 1023;
    const int   hh    = dm >> 6;
    const int   dh    = dm & 63;
#pragma unroll
    for (int v = 0; v < 8; ++v) {
      const int r  = row0 + v + 8 * hlf;   // global token row (C-layout M = v + 8*(lane/16))
      const int bb = r >> 11;
      const int t  = r & (SEQ - 1);
      const float val = accs[nt][v] + bv;
      if (which == 0)
        Qb[((size_t)(bb * NHEADS + hh) * SEQ + t) * HDIM + dh] = f2bf(val * 0.125f); // fold Dh^-0.5
      else if (which == 1)
        Kb[((size_t)(bb * NHEADS + hh) * SEQ + t) * HDIM + dh] = f2bf(val);
      else
        Vb[((size_t)(bb * NHEADS + hh) * HDIM + dh) * SEQ + t] = f2bf(val); // V transposed
    }
  }
}

// ---------------- causal flash attention ----------------
// One wave = 16 query rows; iterate 32-key tiles up to diagonal.

__global__ __launch_bounds__(128) void attn_fa_kernel(
    const unsigned short* __restrict__ Q,   // [B,H,T,Dh] bf16 (pre-scaled)
    const unsigned short* __restrict__ K,   // [B,H,T,Dh] bf16
    const unsigned short* __restrict__ Vt,  // [B,H,Dh,T] bf16
    unsigned short* __restrict__ Ctx) {     // [B,T,D] bf16
  __shared__ alignas(16) unsigned short plds_all[4][16 * 32];
  const int wave  = threadIdx.x >> 5;
  const int lane  = threadIdx.x & 31;
  const int col   = lane & 15;
  const int hlf   = lane >> 4;
  const int qtile = blockIdx.x * 4 + wave;
  const int bh    = blockIdx.y;
  const int b     = bh >> 4;
  const int h     = bh & 15;
  const int qbase = qtile * 16;

  const unsigned short* Qp = Q  + (size_t)bh * SEQ * HDIM;
  const unsigned short* Kp = K  + (size_t)bh * SEQ * HDIM;
  const unsigned short* Vp = Vt + (size_t)bh * HDIM * SEQ;
  unsigned short* plds = plds_all[wave];

  Frag16 qf0, qf1;                       // Q tile 16x64 resident (two K=32 steps)
  loadA(qf0, Qp + (size_t)(qbase + col) * HDIM + 0,  hlf);
  loadA(qf1, Qp + (size_t)(qbase + col) * HDIM + 32, hlf);

  v8f o0 = {}, o1 = {}, o2 = {}, o3 = {};
  float m[8], l[8];
#pragma unroll
  for (int v = 0; v < 8; ++v) { m[v] = -1e30f; l[v] = 0.0f; }

  const int nkt = (qbase + 16 + 31) >> 5;
  for (int kt = 0; kt < nkt; ++kt) {
    const int kb = kt * 32;
    if (kt + 1 < nkt) {                  // pull next K tile rows into cache
      __builtin_prefetch(Kp + (size_t)(kb + 32 + col) * HDIM, 0, 1);
      __builtin_prefetch(Kp + (size_t)(kb + 48 + col) * HDIM, 0, 1);
    }
    v8f s0 = {}, s1 = {};
    {
      Frag16 kf;
      loadB(kf, Kp + (size_t)(kb + col) * HDIM + 0,  hlf);  s0 = wmma_bf16(qf0, kf, s0);
      loadB(kf, Kp + (size_t)(kb + col) * HDIM + 32, hlf);  s0 = wmma_bf16(qf1, kf, s0);
      loadB(kf, Kp + (size_t)(kb + 16 + col) * HDIM + 0,  hlf); s1 = wmma_bf16(qf0, kf, s1);
      loadB(kf, Kp + (size_t)(kb + 16 + col) * HDIM + 32, hlf); s1 = wmma_bf16(qf1, kf, s1);
    }
    // causal mask + online softmax (row r striped over 16-lane half; shfl masks 1..8 stay in-half)
#pragma unroll
    for (int v = 0; v < 8; ++v) {
      const int r = qbase + v + 8 * hlf;
      float a0 = (kb + col      <= r) ? s0[v] : -1e30f;
      float a1 = (kb + 16 + col <= r) ? s1[v] : -1e30f;
      float t = fmaxf(a0, a1);
      t = fmaxf(t, __shfl_xor(t, 1, 32));
      t = fmaxf(t, __shfl_xor(t, 2, 32));
      t = fmaxf(t, __shfl_xor(t, 4, 32));
      t = fmaxf(t, __shfl_xor(t, 8, 32));
      const float nm = fmaxf(m[v], t);
      const float al = __expf(m[v] - nm);
      m[v] = nm;
      const float p0 = __expf(a0 - nm);
      const float p1 = __expf(a1 - nm);
      float sum = p0 + p1;
      sum += __shfl_xor(sum, 1, 32);
      sum += __shfl_xor(sum, 2, 32);
      sum += __shfl_xor(sum, 4, 32);
      sum += __shfl_xor(sum, 8, 32);
      l[v] = l[v] * al + sum;
      const int rr = v + 8 * hlf;
      plds[rr * 32 + col]      = f2bf(p0);   // C-layout -> LDS (row-major 16x32)
      plds[rr * 32 + 16 + col] = f2bf(p1);
      o0[v] *= al; o1[v] *= al; o2[v] *= al; o3[v] *= al;
    }
    // same-wave DS ops are in-order (DScnt); fence keeps compiler from reordering
    asm volatile("s_wait_dscnt 0" ::: "memory");

    Frag16 pf, vf;                       // re-read P as A-fragment (16x32, K = key pos)
    loadA(pf, plds + col * 32, hlf);
    loadB(vf, Vp + (size_t)( 0 + col) * SEQ + kb, hlf); o0 = wmma_bf16(pf, vf, o0);
    loadB(vf, Vp + (size_t)(16 + col) * SEQ + kb, hlf); o1 = wmma_bf16(pf, vf, o1);
    loadB(vf, Vp + (size_t)(32 + col) * SEQ + kb, hlf); o2 = wmma_bf16(pf, vf, o2);
    loadB(vf, Vp + (size_t)(48 + col) * SEQ + kb, hlf); o3 = wmma_bf16(pf, vf, o3);
  }

#pragma unroll
  for (int v = 0; v < 8; ++v) {
    const float inv = 1.0f / l[v];
    const int r = qbase + v + 8 * hlf;
    unsigned short* cp = Ctx + (size_t)(b * SEQ + r) * DMODEL + h * HDIM;
    cp[ 0 + col] = f2bf(o0[v] * inv);
    cp[16 + col] = f2bf(o1[v] * inv);
    cp[32 + col] = f2bf(o2[v] * inv);
    cp[48 + col] = f2bf(o3[v] * inv);
  }
}

// ---------------- output projection GEMM ----------------
// Each wave: 16 rows x 64 cols, A fragment reused 4x per k-step.

__global__ __launch_bounds__(128) void gemm_out_kernel(
    const unsigned short* __restrict__ Cb,   // ctx bf16 [8192,1024]
    const unsigned short* __restrict__ WT,   // Wout^T bf16 [1024,1024]
    const float* __restrict__ bias,          // [1024]
    float* __restrict__ out) {               // fp32 [8192,1024]
  const int wave  = threadIdx.x >> 5;
  const int lane  = threadIdx.x & 31;
  const int col16 = lane & 15;
  const int hlf   = lane >> 4;
  const int row0  = blockIdx.y * 16;
  const int col0  = blockIdx.x * 256 + wave * 64;

  const unsigned short* ap0 = Cb + (size_t)(row0 + col16) * DMODEL;
  const unsigned short* bp0 = WT + (size_t)(col0 + col16) * DMODEL;
  const unsigned short* bp1 = bp0 + (size_t)16 * DMODEL;
  const unsigned short* bp2 = bp0 + (size_t)32 * DMODEL;
  const unsigned short* bp3 = bp0 + (size_t)48 * DMODEL;

  v8f acc0 = {}, acc1 = {}, acc2 = {}, acc3 = {};
  for (int k = 0; k < DMODEL; k += 32) {
    __builtin_prefetch(ap0 + k + 512, 0, 1);
    __builtin_prefetch(bp0 + k + 512, 0, 1);
    Frag16 a, b;
    loadA(a, ap0 + k, hlf);
    loadB(b, bp0 + k, hlf); acc0 = wmma_bf16(a, b, acc0);
    loadB(b, bp1 + k, hlf); acc1 = wmma_bf16(a, b, acc1);
    loadB(b, bp2 + k, hlf); acc2 = wmma_bf16(a, b, acc2);
    loadB(b, bp3 + k, hlf); acc3 = wmma_bf16(a, b, acc3);
  }

  v8f accs[4] = {acc0, acc1, acc2, acc3};
#pragma unroll
  for (int nt = 0; nt < 4; ++nt) {
    const int   c  = col0 + nt * 16 + col16;
    const float bv = bias[c];
#pragma unroll
    for (int v = 0; v < 8; ++v)
      out[(size_t)(row0 + v + 8 * hlf) * DMODEL + c] = accs[nt][v] + bv;
  }
}

// ---------------- launcher ----------------

extern "C" void kernel_launch(void* const* d_in, const int* in_sizes, int n_in,
                              void* d_out, int out_size, void* d_ws, size_t ws_size,
                              hipStream_t stream) {
  (void)in_sizes; (void)n_in; (void)out_size; (void)ws_size;
  const float* x     = (const float*)d_in[0];
  const float* w_qkv = (const float*)d_in[1];
  const float* b_qkv = (const float*)d_in[2];
  const float* w_out = (const float*)d_in[3];
  const float* b_out = (const float*)d_in[4];
  float* out = (float*)d_out;

  char* ws = (char*)d_ws;
  unsigned short* Xb    = (unsigned short*)(ws + 0);                       // 16 MiB
  unsigned short* WqkvT = (unsigned short*)(ws + (size_t)16777216);        //  6 MiB
  unsigned short* WoutT = (unsigned short*)(ws + (size_t)23068672);        //  2 MiB
  unsigned short* Qb    = (unsigned short*)(ws + (size_t)25165824);        // 16 MiB
  unsigned short* Kb    = (unsigned short*)(ws + (size_t)41943040);        // 16 MiB
  unsigned short* Vb    = (unsigned short*)(ws + (size_t)58720256);        // 16 MiB
  unsigned short* Ctx   = (unsigned short*)(ws + (size_t)75497472);        // 16 MiB
  // total workspace use: 92,274,688 bytes

  cvt_bf16_kernel<<<(BT * DMODEL) / 256, 256, 0, stream>>>(x, Xb, BT * DMODEL);
  cvt_t_bf16_kernel<<<(DMODEL * 3 * DMODEL) / 256, 256, 0, stream>>>(w_qkv, WqkvT, DMODEL, 3 * DMODEL);
  cvt_t_bf16_kernel<<<(DMODEL * DMODEL) / 256, 256, 0, stream>>>(w_out, WoutT, DMODEL, DMODEL);

  gemm_qkv_kernel<<<dim3(3 * DMODEL / 256, BT / 16), 128, 0, stream>>>(Xb, WqkvT, b_qkv, Qb, Kb, Vb);

  attn_fa_kernel<<<dim3(SEQ / 16 / 4, BATCH * NHEADS), 128, 0, stream>>>(Qb, Kb, Vb, Ctx);

  gemm_out_kernel<<<dim3(DMODEL / 256, BT / 16), 128, 0, stream>>>(Ctx, WoutT, b_out, out);
}
